// CausalSelfAttention_7988639170735
// MI455X (gfx1250) — compile-verified, matrix operand order checked by reading, not run on hardware
//
#include <hip/hip_runtime.h>
#include <hip/hip_bf16.h>

typedef __attribute__((ext_vector_type(16))) _Float16 v16h;
typedef __attribute__((ext_vector_type(8)))  _Float16 v8h;
typedef __attribute__((ext_vector_type(8)))  float    v8f;
typedef __attribute__((ext_vector_type(4)))  unsigned int u32x4;
typedef __attribute__((ext_vector_type(8)))  int i32x8;
typedef __attribute__((ext_vector_type(4)))  int i32x4;

#define WMMA_F16(A,B,C) __builtin_amdgcn_wmma_f32_16x16x32_f16(false,(A),false,(B),(short)0,(C),false,false)

static constexpr int Bb = 4, Tt = 1024, Cc = 512, Hh = 8, HD = 64;
static constexpr int NROWS = Bb * Tt;            // 4096
static constexpr float SCALE_ = 0.125f;          // 1/sqrt(64)
static constexpr float NEG_BIG = -1.0e30f;
static constexpr int LSTR  = 40;                 // LDS stride (halves) for 64B rows
static constexpr int KSTR  = 72;                 // LDS stride (halves) for 128B K rows

// ---------------------------------------------------------------------------
// fp32 -> f16 convert (grid-stride)
// ---------------------------------------------------------------------------
__global__ void f2h_kernel(const float* __restrict__ src, _Float16* __restrict__ dst, int n) {
    int i = blockIdx.x * blockDim.x + threadIdx.x;
    int stride = gridDim.x * blockDim.x;
    for (; i < n; i += stride) dst[i] = (_Float16)src[i];
}

// ---------------------------------------------------------------------------
// Async copy of one 16B chunk global -> LDS (tracked by ASYNCcnt).
// ---------------------------------------------------------------------------
__device__ __forceinline__ void async_b128(unsigned lds_off, const void* gptr) {
    asm volatile("global_load_async_to_lds_b128 %0, %1, off"
                 :: "v"(lds_off), "v"((unsigned long long)(size_t)gptr)
                 : "memory");
}
__device__ __forceinline__ void wait_async0()  { asm volatile("s_wait_asynccnt 0x0"  ::: "memory"); }
__device__ __forceinline__ void wait_async16() { asm volatile("s_wait_asynccnt 0x10" ::: "memory"); }

// ---------------------------------------------------------------------------
// TDM: DMA a 2D tile (rows x 64B) global -> LDS with 16B row padding
// (LDS row stride = 80B = LSTR halves).  ISA 8.3/8.4 D# layout; 8-byte
// elements; pad_interval=3 (16 DWORDs = one 64B row), pad_amount=3 (4 DWORDs).
// ---------------------------------------------------------------------------
__device__ __forceinline__ void tdm_load_64Brows(unsigned lds_off, const void* gaddr,
                                                 unsigned tile_rows,
                                                 unsigned row_stride_units /*8B units*/) {
    const unsigned long long ga = (unsigned long long)(size_t)gaddr;
    u32x4 g0;
    g0[0] = 1u;                                        // count=1, user mode
    g0[1] = lds_off;                                   // LDS byte address
    g0[2] = (unsigned)(ga & 0xffffffffu);              // global_addr[31:0]
    g0[3] = (unsigned)((ga >> 32) & 0x1ffffffu) | (2u << 30);  // [56:32] | type=2
    i32x8 g1;
    g1[0] = (int)((3u << 16) | (1u << 20) | (3u << 22) | (3u << 25)); // dsz=8B,pad
    g1[1] = (int)((row_stride_units & 0xffffu) << 16); // tensor_dim0 lo (use stride)
    g1[2] = (int)(((row_stride_units >> 16) & 0xffffu) | ((tile_rows & 0xffffu) << 16)); // dim0 hi | dim1 lo
    g1[3] = (int)(((tile_rows >> 16) & 0xffffu) | (8u << 16));  // dim1 hi | tile_dim0=8 (64B)
    g1[4] = (int)(tile_rows & 0xffffu);                // tile_dim1 | tile_dim2=0
    g1[5] = (int)row_stride_units;                     // tensor_dim0_stride lo
    g1[6] = 0; g1[7] = 0;
    const i32x4 z4 = {0, 0, 0, 0};
#if __clang_major__ >= 23
    const i32x8 z8 = {0, 0, 0, 0, 0, 0, 0, 0};
    __builtin_amdgcn_tensor_load_to_lds(g0, g1, z4, z4, z8, 0);
#else
    __builtin_amdgcn_tensor_load_to_lds(g0, g1, z4, z4, 0);
#endif
}

// ---------------------------------------------------------------------------
// Load a 16x32 f16 A/B fragment chunk pair for this lane.
// Per ISA 7.12.2: halves 0..7 -> K = 8*hi..8*hi+7, halves 8..15 -> +16.
// ---------------------------------------------------------------------------
__device__ __forceinline__ v16h load_frag(const _Float16* base, int hi) {
    const v8h lo = *(const v8h*)(base + 8 * hi);
    const v8h hp = *(const v8h*)(base + 16 + 8 * hi);
    v16h r;
#pragma unroll
    for (int i = 0; i < 8; ++i) { r[i] = lo[i]; r[i + 8] = hp[i]; }
    return r;
}

// ---------------------------------------------------------------------------
// QKV projection: out(4096x512) = X(f16) @ W^T(f16) + bias.
// Block = 8 waves, one 16-row m-tile; A/B K-slices staged double-buffered in
// LDS by the Tensor Data Mover (wave 0 issues, block barriers), WMMA from LDS.
// store_mode 0: out[((b*H+h)*T + t)*64 + d];  1: out[((b*H+h)*64 + d)*T + t].
// ---------------------------------------------------------------------------
__global__ void gemm_qkv_kernel(const _Float16* __restrict__ X,
                                const _Float16* __restrict__ W,
                                const float* __restrict__ bias,
                                _Float16* __restrict__ out, int store_mode) {
    __shared__ alignas(16) _Float16 As[2][16 * LSTR];
    __shared__ alignas(16) _Float16 Bs[2][512 * LSTR];

    const int tid  = threadIdx.x;
    const int warp = tid >> 5, lane = tid & 31;
    const int ln = lane & 15, hi = lane >> 4;
    const int m0 = blockIdx.x * 16;
    const int n0 = warp * 64;

    auto issueTile = [&](int buf, int k0) {
        tdm_load_64Brows((unsigned)(size_t)(&Bs[buf][0]), W + k0, 512u, (unsigned)Cc / 4u);
        tdm_load_64Brows((unsigned)(size_t)(&As[buf][0]), X + (long)m0 * Cc + k0, 16u, (unsigned)Cc / 4u);
    };

    const v8f zero = {0.f, 0.f, 0.f, 0.f, 0.f, 0.f, 0.f, 0.f};
    v8f acc[4];
#pragma unroll
    for (int nn = 0; nn < 4; ++nn) acc[nn] = zero;

    if (warp == 0) issueTile(0, 0);
    int buf = 0;
    for (int k0 = 0; k0 < Cc; k0 += 32) {
        if (warp == 0) __builtin_amdgcn_s_wait_tensorcnt(0);
        __syncthreads();        // DMA visible to all; prev compute finished
        if (warp == 0 && k0 + 32 < Cc) issueTile(buf ^ 1, k0 + 32);

        const v16h a = load_frag(&As[buf][ln * LSTR], hi);
#pragma unroll
        for (int nn = 0; nn < 4; ++nn) {
            const v16h b = load_frag(&Bs[buf][(n0 + nn * 16 + ln) * LSTR], hi);
            acc[nn] = WMMA_F16(a, b, acc[nn]);
        }
        buf ^= 1;
    }

#pragma unroll
    for (int nn = 0; nn < 4; ++nn) {
        const int o = n0 + nn * 16 + ln;       // output channel
        const int h = o >> 6, d = o & 63;
        const float bvv = bias[o];
#pragma unroll
        for (int j = 0; j < 8; ++j) {
            const int g = m0 + j + 8 * hi;     // global token row
            const int bb = g >> 10, t = g & 1023;
            const float val = acc[nn][j] + bvv;
            long idx;
            if (store_mode) idx = ((long)(bb * Hh + h) * HD + d) * Tt + t;
            else            idx = ((long)(bb * Hh + h) * Tt + t) * HD + d;
            out[idx] = (_Float16)val;
        }
    }
}

// ---------------------------------------------------------------------------
// Flash attention: one wave per (b, h, 16-query tile).  K/V tiles are
// double-buffered in LDS via async global->LDS copies (16 per lane per tile);
// s_wait_asynccnt 16 after issuing the next tile => previous tile landed.
// ---------------------------------------------------------------------------
__global__ void attn_kernel(const _Float16* __restrict__ Q,
                            const _Float16* __restrict__ K,
                            const _Float16* __restrict__ VT,   // [bh][d][t]
                            const float* __restrict__ attn_bias,
                            _Float16* __restrict__ Y) {
    __shared__ alignas(16) _Float16 P[16 * LSTR];
    __shared__ alignas(16) _Float16 Ks[2][32 * KSTR];
    __shared__ alignas(16) _Float16 Vs[2][64 * LSTR];

    const int lane = threadIdx.x & 31;
    const int ln = lane & 15, hi = lane >> 4;
    const int wid = blockIdx.x;                // 0..2047
    const int qt = wid & 63;                   // query tile (16 rows)
    const int bh = wid >> 6;                   // 0..31
    const int h  = bh & 7, bidx = bh >> 3;

    const _Float16* qb  = Q  + ((long)bh * Tt + qt * 16) * HD;
    const _Float16* kb  = K  + (long)bh * Tt * HD;
    const _Float16* vtb = VT + (long)bh * HD * Tt;

    // stage 32-key K tile (rows: [key][dim], stride KSTR) and V tile
    // (rows: [dim][key], stride LSTR): 8 + 8 async b128 copies per lane.
    auto issueKV = [&](int buf, int kt) {
        const _Float16* ksrc = kb + (long)(kt + lane) * HD;
        unsigned kdst = (unsigned)(size_t)(&Ks[buf][lane * KSTR]);
#pragma unroll
        for (int c = 0; c < 8; ++c) async_b128(kdst + 16 * c, ksrc + 8 * c);
#pragma unroll
        for (int rr = 0; rr < 2; ++rr) {
            const int d = lane + 32 * rr;
            const _Float16* vsrc = vtb + (long)d * Tt + kt;
            unsigned vdst = (unsigned)(size_t)(&Vs[buf][d * LSTR]);
#pragma unroll
            for (int c = 0; c < 4; ++c) async_b128(vdst + 16 * c, vsrc + 8 * c);
        }
    };

    const v16h qa0 = load_frag(qb + ln * HD,      hi);   // k-dim 0..31
    const v16h qa1 = load_frag(qb + ln * HD + 32, hi);   // k-dim 32..63

    const v8f zero = {0.f, 0.f, 0.f, 0.f, 0.f, 0.f, 0.f, 0.f};
    v8f oacc[4];
#pragma unroll
    for (int nn = 0; nn < 4; ++nn) oacc[nn] = zero;
    float mrow[8], lrow[8];
#pragma unroll
    for (int j = 0; j < 8; ++j) { mrow[j] = NEG_BIG; lrow[j] = 0.f; }

    const int qhi  = qt * 16 + 15;
    const int kend = ((qhi >> 5) + 1) << 5;    // keys needed, rounded to 32

    issueKV(0, 0);
    int buf = 0;
    for (int kt = 0; kt < kend; kt += 32) {
        if (kt + 32 < kend) { issueKV(buf ^ 1, kt + 32); wait_async16(); }
        else                { wait_async0(); }

        float sv[2][8];
#pragma unroll
        for (int f = 0; f < 2; ++f) {
            const int tk = kt + 16 * f + ln;   // this lane's key column
            const _Float16* krow = &Ks[buf][(16 * f + ln) * KSTR];
            v16h b0 = load_frag(krow,      hi);
            v16h b1 = load_frag(krow + 32, hi);
            v8f s = WMMA_F16(qa0, b0, zero);
            s = WMMA_F16(qa1, b1, s);
#pragma unroll
            for (int j = 0; j < 8; ++j) {
                const int tq = qt * 16 + j + 8 * hi;
                float val = s[j] * SCALE_;
                if (tq < 512 && tk < 512)
                    val += attn_bias[h * 4096 + (tq >> 3) * 64 + (tk >> 3)];
                if (tk > tq || (tk & 15) == 15) val = NEG_BIG;  // causal + col mask
                sv[f][j] = val;
            }
        }

        // online softmax update; rows live in 16-lane groups of the D layout
#pragma unroll
        for (int j = 0; j < 8; ++j) {
            float rm = fmaxf(sv[0][j], sv[1][j]);
#pragma unroll
            for (int off = 1; off < 16; off <<= 1)
                rm = fmaxf(rm, __shfl_xor(rm, off, 32));
            const float mnew  = fmaxf(mrow[j], rm);
            const float alpha = __expf(mrow[j] - mnew);
            const float p0 = __expf(sv[0][j] - mnew);
            const float p1 = __expf(sv[1][j] - mnew);
            float rs = p0 + p1;
#pragma unroll
            for (int off = 1; off < 16; off <<= 1)
                rs += __shfl_xor(rs, off, 32);
            lrow[j] = lrow[j] * alpha + rs;
            mrow[j] = mnew;
#pragma unroll
            for (int nn = 0; nn < 4; ++nn) oacc[nn][j] *= alpha;
            const int m = j + 8 * hi;
            P[m * LSTR + ln]      = (_Float16)p0;
            P[m * LSTR + 16 + ln] = (_Float16)p1;
        }
        asm volatile("s_wait_dscnt 0x0" ::: "memory");

        // P (16x32) as A-fragment; V tile B-frags from LDS
        const v16h pa = load_frag(&P[ln * LSTR], hi);
#pragma unroll
        for (int nn = 0; nn < 4; ++nn) {
            const v16h vb = load_frag(&Vs[buf][(nn * 16 + ln) * LSTR], hi);
            oacc[nn] = WMMA_F16(pa, vb, oacc[nn]);
        }
        asm volatile("s_wait_dscnt 0x0" ::: "memory");
        buf ^= 1;
    }

    // epilogue: normalize and store y[b][t][h*64+d] as f16
#pragma unroll
    for (int j = 0; j < 8; ++j) {
        const float inv = 1.0f / lrow[j];
        const int t = qt * 16 + j + 8 * hi;
        const long rowbase = ((long)bidx * Tt + t) * Cc + h * HD;
#pragma unroll
        for (int nn = 0; nn < 4; ++nn)
            Y[rowbase + nn * 16 + ln] = (_Float16)(oacc[nn][j] * inv);
    }
}

// ---------------------------------------------------------------------------
// Output projection: out(4096x512,f32) = Y(f16) @ Wp^T + bp (TDM pipeline)
// ---------------------------------------------------------------------------
__global__ void gemm_out_kernel(const _Float16* __restrict__ Yh,
                                const _Float16* __restrict__ W,
                                const float* __restrict__ bias,
                                float* __restrict__ out) {
    __shared__ alignas(16) _Float16 As[2][16 * LSTR];
    __shared__ alignas(16) _Float16 Bs[2][512 * LSTR];

    const int tid  = threadIdx.x;
    const int warp = tid >> 5, lane = tid & 31;
    const int ln = lane & 15, hi = lane >> 4;
    const int m0 = blockIdx.x * 16;
    const int n0 = warp * 64;

    auto issueTile = [&](int buf, int k0) {
        tdm_load_64Brows((unsigned)(size_t)(&Bs[buf][0]), W + k0, 512u, (unsigned)Cc / 4u);
        tdm_load_64Brows((unsigned)(size_t)(&As[buf][0]), Yh + (long)m0 * Cc + k0, 16u, (unsigned)Cc / 4u);
    };

    const v8f zero = {0.f, 0.f, 0.f, 0.f, 0.f, 0.f, 0.f, 0.f};
    v8f acc[4];
#pragma unroll
    for (int nn = 0; nn < 4; ++nn) acc[nn] = zero;

    if (warp == 0) issueTile(0, 0);
    int buf = 0;
    for (int k0 = 0; k0 < Cc; k0 += 32) {
        if (warp == 0) __builtin_amdgcn_s_wait_tensorcnt(0);
        __syncthreads();
        if (warp == 0 && k0 + 32 < Cc) issueTile(buf ^ 1, k0 + 32);

        const v16h a = load_frag(&As[buf][ln * LSTR], hi);
#pragma unroll
        for (int nn = 0; nn < 4; ++nn) {
            const v16h b = load_frag(&Bs[buf][(n0 + nn * 16 + ln) * LSTR], hi);
            acc[nn] = WMMA_F16(a, b, acc[nn]);
        }
        buf ^= 1;
    }

#pragma unroll
    for (int nn = 0; nn < 4; ++nn) {
        const int o = n0 + nn * 16 + ln;
        const float bvv = bias[o];
#pragma unroll
        for (int j = 0; j < 8; ++j) {
            const int g = m0 + j + 8 * hi;
            out[(long)g * Cc + o] = acc[nn][j] + bvv;
        }
    }
}

// ---------------------------------------------------------------------------
extern "C" void kernel_launch(void* const* d_in, const int* in_sizes, int n_in,
                              void* d_out, int out_size, void* d_ws, size_t ws_size,
                              hipStream_t stream) {
    const float* x  = (const float*)d_in[0];
    const float* ab = (const float*)d_in[1];
    const float* Wq = (const float*)d_in[2];
    const float* bq = (const float*)d_in[3];
    const float* Wk = (const float*)d_in[4];
    const float* bk = (const float*)d_in[5];
    const float* Wv = (const float*)d_in[6];
    const float* bv = (const float*)d_in[7];
    const float* Wp = (const float*)d_in[8];
    const float* bp = (const float*)d_in[9];
    float* out = (float*)d_out;
    char* ws = (char*)d_ws;

    const size_t SX = (size_t)NROWS * Cc;   // 2,097,152 elems
    const size_t SW = (size_t)Cc * Cc;      //   262,144 elems

    _Float16* xh  = (_Float16*)(ws);
    _Float16* wqh = (_Float16*)(ws + SX * 2);
    _Float16* wkh = (_Float16*)(ws + SX * 2 + SW * 2);
    _Float16* wvh = (_Float16*)(ws + SX * 2 + SW * 4);
    _Float16* wph = (_Float16*)(ws + SX * 2 + SW * 6);
    _Float16* qh  = (_Float16*)(ws + SX * 2 + SW * 8);
    _Float16* kh  = (_Float16*)(ws + SX * 4 + SW * 8);
    _Float16* vth = (_Float16*)(ws + SX * 6 + SW * 8);
    _Float16* yh  = (_Float16*)(ws + SX * 8 + SW * 8);

    f2h_kernel<<<1024, 256, 0, stream>>>(x,  xh,  (int)SX);
    f2h_kernel<<<256,  256, 0, stream>>>(Wq, wqh, (int)SW);
    f2h_kernel<<<256,  256, 0, stream>>>(Wk, wkh, (int)SW);
    f2h_kernel<<<256,  256, 0, stream>>>(Wv, wvh, (int)SW);
    f2h_kernel<<<256,  256, 0, stream>>>(Wp, wph, (int)SW);

    // one block (8 waves) per 16-row m-tile; waves cover the 512 output cols
    gemm_qkv_kernel<<<256, 256, 0, stream>>>(xh, wqh, bq, qh,  0);
    gemm_qkv_kernel<<<256, 256, 0, stream>>>(xh, wkh, bk, kh,  0);
    gemm_qkv_kernel<<<256, 256, 0, stream>>>(xh, wvh, bv, vth, 1);

    // one wave per (b,h,qtile): 32 * 64 = 2048 blocks of one wave
    attn_kernel<<<2048, 32, 0, stream>>>(qh, kh, vth, ab, yh);

    gemm_out_kernel<<<256, 256, 0, stream>>>(yh, wph, bp, out);
}
